// SwinBlock_3221225472069
// MI455X (gfx1250) — compile-verified, matrix operand order checked by reading, not run on hardware
//
#include <hip/hip_runtime.h>
#include <math.h>

typedef __attribute__((ext_vector_type(16))) __bf16        v16bf;
typedef __attribute__((ext_vector_type(8)))  float         v8f;
typedef __attribute__((ext_vector_type(4)))  unsigned int  u32x4;

#define C    192
#define T    64
#define NH   4
#define HD   48
#define HDP  64
#define MH   768
#define NTHREADS 256

// ---------- gfx1250 async-to-LDS gather (guarded; falls back to plain loads) ----------
#if defined(__gfx1250__) && __has_builtin(__builtin_amdgcn_global_load_async_to_lds_b32)
#define USE_ASYNC_GATHER 1
#if __has_builtin(__builtin_amdgcn_s_wait_asynccnt)
#define WAIT_ASYNC() __builtin_amdgcn_s_wait_asynccnt(0)
#else
#define WAIT_ASYNC() asm volatile("s_wait_asynccnt 0x0" ::: "memory")
#endif
typedef __attribute__((address_space(1))) int* gas1_int;   // global ptr (prints as __device__ int*)
typedef __attribute__((address_space(3))) int* las3_int;   // LDS ptr
#endif

// ---------- helpers ----------
__device__ __forceinline__ unsigned short f2bf(float f) {
  unsigned int x = __builtin_bit_cast(unsigned int, f);
  unsigned int r = x + 0x7FFFu + ((x >> 16) & 1u);   // round-to-nearest-even
  return (unsigned short)(r >> 16);
}

__device__ __forceinline__ v16bf make_frag(u32x4 lo, u32x4 hi) {
  union { u32x4 q[2]; v16bf v; } u;
  u.q[0] = lo; u.q[1] = hi;
  return u.v;
}

// Row-major bf16 matrix, one fragment = 16 bf16 per lane (two b128 loads).
// A 16x32 per ISA: lane m=L%16, half=L/16; K-pairs contiguous.
// B (32x16) uses the same loader on the [N][K]-stored matrix (B[k][n]=Bt[n][k]).
__device__ __forceinline__ v16bf load_frag(const unsigned short* base, int row,
                                           int stride, int k0, int half) {
  const u32x4* p = (const u32x4*)(base + row * stride + k0);
  return make_frag(p[half], p[2 + half]);
}

__device__ __forceinline__ v8f wmma_bf16(v16bf a, v16bf b, v8f c) {
  return __builtin_amdgcn_wmma_f32_16x16x32_bf16(false, a, false, b,
                                                 (short)0, c, false, false);
}

// One column tile (16 cols), all 4 row tiles (64 rows), K = KT*32 starting at kbase.
// B fragments are held in registers and reused across the 4 M tiles.
template <int KT>
__device__ __forceinline__ void gemm_cols(const unsigned short* __restrict__ A,
                                          const unsigned short* __restrict__ Bt,
                                          int stride, int kbase,
                                          int nt, int l16, int half, v8f* acc) {
  v16bf bfr[KT];
#pragma unroll
  for (int kt = 0; kt < KT; ++kt)
    bfr[kt] = load_frag(Bt, nt * 16 + l16, stride, kbase + kt * 32, half);
#pragma unroll
  for (int mt = 0; mt < 4; ++mt) {
#pragma unroll
    for (int kt = 0; kt < KT; ++kt) {
      v16bf a = load_frag(A, mt * 16 + l16, stride, kbase + kt * 32, half);
      acc[mt] = wmma_bf16(a, bfr[kt], acc[mt]);
    }
  }
}

// ---------- weight pre-transpose: W[K][N] f32 -> Wt[N][K] bf16 ----------
__global__ __launch_bounds__(256)
void conv_w(const float* __restrict__ W, unsigned short* __restrict__ Wt,
            int K, int N) {
  int i = blockIdx.x * blockDim.x + threadIdx.x;
  if (i >= K * N) return;
  int n = i / K, k = i % K;
  Wt[i] = f2bf(W[k * N + n]);
}

// ---------- fused Swin block: one workgroup (8 wave32) per 8x8 window ----------
__global__ __launch_bounds__(256)
void swin_block_kernel(const float* __restrict__ x,
                       const float* __restrict__ g1, const float* __restrict__ b1,
                       const unsigned short* __restrict__ wqkvT, const float* __restrict__ bqkv,
                       const unsigned short* __restrict__ woT,   const float* __restrict__ bo,
                       const float* __restrict__ g2, const float* __restrict__ b2,
                       const unsigned short* __restrict__ wm1T,  const float* __restrict__ bm1,
                       const unsigned short* __restrict__ wm2T,  const float* __restrict__ bm2,
                       float* __restrict__ out) {
  extern __shared__ char smem[];
  float*          Xres  = (float*)(smem);                    // [64][192] f32 residual
  unsigned short* Xn    = (unsigned short*)(smem + 49152);   // [64][192] bf16 (LN out)
  unsigned short* Qs    = (unsigned short*)(smem + 73728);   // [4][64][64] bf16, d-padded
  unsigned short* Ks    = (unsigned short*)(smem + 106496);  // [4][64][64]
  unsigned short* Vt    = (unsigned short*)(smem + 139264);  // [4][64(d)][64(tok)]
  unsigned short* Gb    = (unsigned short*)(smem + 73728);   // [64][768] aliases Q/K/V
  float*          Sm4   = (float*)(smem + 172032);           // [4][64][64] scores f32
  unsigned short* Pb4   = (unsigned short*)(smem + 237568);  // [4][64][64] probs bf16
  unsigned short* Ob    = (unsigned short*)(smem + 270336);  // [64][192] attn out bf16
  float*          stats = (float*)(smem + 294912);           // [128] mu / rstd

  const int tid  = threadIdx.x;
  const int wave = tid >> 5;
  const int lane = tid & 31;
  const int half = lane >> 4;
  const int l16  = lane & 15;

  const int wi = blockIdx.x;          // window id: 4096 = 4 * 32 * 32
  const int bb = wi >> 10;
  const int wy = (wi >> 5) & 31;
  const int wx = wi & 31;

  // ---- L2 prefetch of the bf16 weight pool (first workgroups only) ----
  if (wi < 256) {
    const char* wp = (const char*)wqkvT;        // all 4 weights contiguous: 884736 B
    for (int i = tid * 256; i < 884736; i += NTHREADS * 256)
      __builtin_prefetch(wp + i, 0, 1);
  }

  // ---- phase 0: shifted gather (roll by -4) + zero padded Q/K/V ----
#if USE_ASYNC_GATHER
  for (int i = tid; i < T * C; i += NTHREADS) {
    int t = i & 63, c = i >> 6;
    int ty = t >> 3, tx = t & 7;
    int h = (wy * 8 + ty + 4) & 255;
    int w = (wx * 8 + tx + 4) & 255;
    const float* src = x + (((bb * C + c) << 16) + (h << 8) + w);
    float*       dst = &Xres[t * C + c];
    __builtin_amdgcn_global_load_async_to_lds_b32((gas1_int)src, (las3_int)dst, 0, 0);
  }
#else
  for (int i = tid; i < T * C; i += NTHREADS) {
    int t = i & 63, c = i >> 6;
    int ty = t >> 3, tx = t & 7;
    int h = (wy * 8 + ty + 4) & 255;
    int w = (wx * 8 + tx + 4) & 255;
    Xres[t * C + c] = x[((bb * C + c) << 16) + (h << 8) + w];
  }
#endif
  {
    unsigned int* z = (unsigned int*)Qs;        // zero Q/K/V region (98304 B)
    for (int i = tid; i < 24576; i += NTHREADS) z[i] = 0u;
  }
#if USE_ASYNC_GATHER
  WAIT_ASYNC();
#endif
  __syncthreads();

  // ---- phase 1: LayerNorm 1 -> Xn (bf16) ----
  if (tid < T) {
    float s = 0.f, s2 = 0.f;
    for (int c = 0; c < C; ++c) { float v = Xres[tid * C + c]; s += v; s2 += v * v; }
    float mu  = s * (1.0f / C);
    float var = s2 * (1.0f / C) - mu * mu;
    stats[tid]      = mu;
    stats[64 + tid] = rsqrtf(var + 1e-5f);
  }
  __syncthreads();
  for (int i = tid; i < T * C; i += NTHREADS) {
    int t = i / C, c = i % C;
    float v = (Xres[i] - stats[t]) * stats[64 + t] * g1[c] + b1[c];
    Xn[i] = f2bf(v);
  }
  __syncthreads();

  // ---- phase 2: QKV GEMM (64x192)x(192x576); B frags register-resident ----
  const float scale = 0.14433756729740643f;     // 1/sqrt(48)
  for (int nt = wave; nt < 36; nt += 8) {
    v8f acc[4] = {};
    gemm_cols<6>(Xn, wqkvT, C, 0, nt, l16, half, acc);
    int n = nt * 16 + l16;
    float bias = bqkv[n];
    int seg = n / C;                 // 0=Q 1=K 2=V (uniform per tile: 192 % 16 == 0)
    int cc = n - seg * C;
    int hh = cc / HD, d = cc - hh * HD;
#pragma unroll
    for (int mt = 0; mt < 4; ++mt) {
#pragma unroll
      for (int r = 0; r < 8; ++r) {
        int m = mt * 16 + r + 8 * half;            // token
        float v = acc[mt][r] + bias;
        if (seg == 0)      Qs[(hh * T + m) * HDP + d] = f2bf(v * scale);
        else if (seg == 1) Ks[(hh * T + m) * HDP + d] = f2bf(v);
        else               Vt[(hh * T + d) * HDP + m] = f2bf(v);
      }
    }
  }
  __syncthreads();

  // ---- phase 3a: S = Q * K^T for all heads (16 column units across 8 waves) ----
  for (int u = wave; u < 16; u += 8) {
    int hh = u >> 2, nt = u & 3;
    v8f acc[4] = {};
    gemm_cols<2>(Qs + hh * T * HDP, Ks + hh * T * HDP, HDP, 0, nt, l16, half, acc);
    float* Sh = Sm4 + hh * T * T;
#pragma unroll
    for (int mt = 0; mt < 4; ++mt)
#pragma unroll
      for (int r = 0; r < 8; ++r)
        Sh[(mt * 16 + r + 8 * half) * T + nt * 16 + l16] = acc[mt][r];
  }
  __syncthreads();

  // ---- phase 3b: softmax, one row per thread (4 heads * 64 rows = 256) ----
  {
    float* row = Sm4 + tid * T;                   // tid = hh*64 + r
    float mx = -3.0e38f;
    for (int j = 0; j < T; ++j) mx = fmaxf(mx, row[j]);
    float s = 0.f;
    for (int j = 0; j < T; ++j) { float e = __expf(row[j] - mx); row[j] = e; s += e; }
    float inv = 1.0f / s;
    unsigned short* prow = Pb4 + tid * T;
    for (int j = 0; j < T; ++j) prow[j] = f2bf(row[j] * inv);
  }
  __syncthreads();

  // ---- phase 3c: O = P * V for all heads (12 column units) ----
  for (int u = wave; u < 12; u += 8) {
    int hh = u / 3, nt = u % 3;
    v8f acc[4] = {};
    gemm_cols<2>(Pb4 + hh * T * T, Vt + hh * T * HDP, T /*==HDP*/, 0, nt, l16, half, acc);
#pragma unroll
    for (int mt = 0; mt < 4; ++mt)
#pragma unroll
      for (int r = 0; r < 8; ++r) {
        int m = mt * 16 + r + 8 * half;
        Ob[m * C + hh * HD + nt * 16 + l16] = f2bf(acc[mt][r]);
      }
  }
  __syncthreads();

  // ---- phase 4: output projection + residual ----
  for (int nt = wave; nt < 12; nt += 8) {
    v8f acc[4] = {};
    gemm_cols<6>(Ob, woT, C, 0, nt, l16, half, acc);
    int n = nt * 16 + l16;
    float bias = bo[n];
#pragma unroll
    for (int mt = 0; mt < 4; ++mt)
#pragma unroll
      for (int r = 0; r < 8; ++r) {
        int m = mt * 16 + r + 8 * half;
        Xres[m * C + n] += acc[mt][r] + bias;
      }
  }
  __syncthreads();

  // ---- phase 5: LayerNorm 2 -> Xn ----
  if (tid < T) {
    float s = 0.f, s2 = 0.f;
    for (int c = 0; c < C; ++c) { float v = Xres[tid * C + c]; s += v; s2 += v * v; }
    float mu  = s * (1.0f / C);
    float var = s2 * (1.0f / C) - mu * mu;
    stats[tid]      = mu;
    stats[64 + tid] = rsqrtf(var + 1e-5f);
  }
  __syncthreads();
  for (int i = tid; i < T * C; i += NTHREADS) {
    int t = i / C, c = i % C;
    float v = (Xres[i] - stats[t]) * stats[64 + t] * g2[c] + b2[c];
    Xn[i] = f2bf(v);
  }
  __syncthreads();

  // ---- phase 6: MLP fc1 + exact GELU -> Gb (aliases dead Q/K/V) ----
  for (int nt = wave; nt < 48; nt += 8) {
    v8f acc[4] = {};
    gemm_cols<6>(Xn, wm1T, C, 0, nt, l16, half, acc);
    int n = nt * 16 + l16;
    float bias = bm1[n];
#pragma unroll
    for (int mt = 0; mt < 4; ++mt)
#pragma unroll
      for (int r = 0; r < 8; ++r) {
        int m = mt * 16 + r + 8 * half;
        float v = acc[mt][r] + bias;
        v = 0.5f * v * (1.0f + erff(v * 0.70710678118654752f));
        Gb[m * MH + n] = f2bf(v);
      }
  }
  __syncthreads();

  // ---- phase 7: MLP fc2 + residual + shifted scatter (roll by +4) ----
  for (int nt = wave; nt < 12; nt += 8) {
    v8f acc[4] = {};
#pragma unroll
    for (int kc = 0; kc < 4; ++kc)
      gemm_cols<6>(Gb, wm2T, MH, kc * 192, nt, l16, half, acc);
    int n = nt * 16 + l16;
    float bias = bm2[n];
#pragma unroll
    for (int mt = 0; mt < 4; ++mt)
#pragma unroll
      for (int r = 0; r < 8; ++r) {
        int m = mt * 16 + r + 8 * half;
        float v = Xres[m * C + n] + acc[mt][r] + bias;
        int ty = m >> 3, tx = m & 7;
        int h = (wy * 8 + ty + 4) & 255;
        int w = (wx * 8 + tx + 4) & 255;
        out[((bb * C + n) << 16) + (h << 8) + w] = v;
      }
  }
}

// ---------- host launcher ----------
extern "C" void kernel_launch(void* const* d_in, const int* in_sizes, int n_in,
                              void* d_out, int out_size, void* d_ws, size_t ws_size,
                              hipStream_t stream) {
  const float* x    = (const float*)d_in[0];
  const float* g1   = (const float*)d_in[1];
  const float* b1   = (const float*)d_in[2];
  const float* wqkv = (const float*)d_in[3];
  const float* bqkv = (const float*)d_in[4];
  const float* wo   = (const float*)d_in[5];
  const float* bo   = (const float*)d_in[6];
  const float* g2   = (const float*)d_in[7];
  const float* b2   = (const float*)d_in[8];
  const float* wm1  = (const float*)d_in[9];
  const float* bm1  = (const float*)d_in[10];
  const float* wm2  = (const float*)d_in[11];
  const float* bm2  = (const float*)d_in[12];
  float* out = (float*)d_out;

  unsigned short* ws    = (unsigned short*)d_ws;
  unsigned short* wqkvT = ws;                 // [576][192] bf16
  unsigned short* woT   = ws + 110592;        // [192][192]
  unsigned short* wm1T  = ws + 147456;        // [768][192]
  unsigned short* wm2T  = ws + 294912;        // [192][768]

  conv_w<<<(192 * 576 + 255) / 256, 256, 0, stream>>>(wqkv, wqkvT, 192, 576);
  conv_w<<<(192 * 192 + 255) / 256, 256, 0, stream>>>(wo,   woT,   192, 192);
  conv_w<<<(192 * 768 + 255) / 256, 256, 0, stream>>>(wm1,  wm1T,  192, 768);
  conv_w<<<(768 * 192 + 255) / 256, 256, 0, stream>>>(wm2,  wm2T,  768, 192);

  const size_t smem = 295424;                 // ~289 KB of 320 KB per WGP
  swin_block_kernel<<<4096, 256, smem, stream>>>(
      x, g1, b1, wqkvT, bqkv, woT, bo, g2, b2, wm1T, bm1, wm2T, bm2, out);
}
